// SpatialHRVQTokenizer_57080115364778
// MI455X (gfx1250) — matrix operand order repaired and checked by reading, not compiled, not run on hardware
//
#include <hip/hip_runtime.h>
#include <hip/hip_bf16.h>

#ifndef VQ_USE_ASYNC_LDS
#define VQ_USE_ASYNC_LDS 1   // 1: global_load_async_to_lds_b128 + s_wait_asynccnt
#endif

// ---------- types ----------
typedef __attribute__((ext_vector_type(16))) __bf16       v16bf;
typedef __attribute__((ext_vector_type(8)))  float        v8f;
typedef __attribute__((ext_vector_type(8)))  unsigned int v8u;

struct B32 { uint4 lo; uint4 hi; };   // 32 bytes = 16 bf16

// float -> bf16 bits, round to nearest even
__device__ __forceinline__ unsigned short f2bf(float f) {
    unsigned int u = __float_as_uint(f);
    unsigned int r = u + 0x7FFFu + ((u >> 16) & 1u);
    return (unsigned short)(r >> 16);
}
__device__ __forceinline__ unsigned int pk(float lo, float hi) {
    return (unsigned int)f2bf(lo) | ((unsigned int)f2bf(hi) << 16);
}

// ---------- pre-pass: fp32 codebook -> bf16 bits ----------
__global__ void cvt_bf16_kernel(const float* __restrict__ in,
                                unsigned short* __restrict__ out, int n) {
    int i = blockIdx.x * 256 + threadIdx.x;
    if (i < n) out[i] = f2bf(in[i]);
}

// ---------- pre-pass: ||cb_k||^2 in fp32 ----------
__global__ void norm_kernel(const float* __restrict__ cb,
                            float* __restrict__ nrm, int K) {
    int k = blockIdx.x * 256 + threadIdx.x;
    if (k < K) {
        const float* r = cb + (size_t)k * 384;
        float s = 0.f;
        #pragma unroll 4
        for (int i = 0; i < 384; ++i) s = fmaf(r[i], r[i], s);
        nrm[k] = s;
    }
}

__global__ void init_acc_kernel(float* acc) {
    if (threadIdx.x < 3) acc[threadIdx.x] = 0.f;
}

// ---------- WMMA argmin kernel with async LDS double-buffered codebook ----------
// block = 256 threads = 8 waves; each wave owns a 16-row tile of x (128 rows/block).
// Codebook staged 32 codes at a time into LDS (one copy per BLOCK, async DMA),
// double-buffered so tile ct+1 streams in while the 24 WMMAs of tile ct run.
// score[n,k] = ||cb_k||^2 - 2 * dot(x_n, cb_k)   (||x||^2 constant per row)
#define LROW 392                 // LDS row stride in ushorts (768B data + 16B pad)
#define SBELEM (32 * LROW)       // elements per stage buffer

__global__ __launch_bounds__(256)
void vq_argmin_kernel(const float* __restrict__ x,
                      const unsigned short* __restrict__ cbf,  // bf16 bits [K,384]
                      const float* __restrict__ cbnorm,        // [K]
                      float* __restrict__ idxf_out,            // [N] idx as float
                      int K) {
    __shared__ __align__(16) unsigned short sbuf[2 * SBELEM];  // 50,176 B

    const int tid  = threadIdx.x;
    const int lane = tid & 31;
    const int wave = tid >> 5;
    const int lr   = lane & 15;   // row (A) / code column (B,C)
    const int half = lane >> 4;   // K-half selector
    const long row0 = (long)blockIdx.x * 128 + (long)wave * 16;

    // stage one 32-code tile (32 x 768B = 24KB) into sbuf[sb..]
    // 256 threads: row = tid&31, 8 threads per row, 6 x 16B chunks each.
    auto stage = [&](int cbase, int sb) {
        const int row = tid & 31;
        const int j   = tid >> 5;                       // 0..7
        const unsigned short* g = cbf + (size_t)(cbase + row) * 384 + j * 8;
#if VQ_USE_ASYNC_LDS
        unsigned l = (unsigned)(unsigned long long)&sbuf[sb + row * LROW + j * 8];
        #pragma unroll
        for (int i = 0; i < 6; ++i) {
            unsigned long long ga = (unsigned long long)(g + i * 64);
            asm volatile("global_load_async_to_lds_b128 %0, %1, off"
                         :: "v"(l + i * 128), "v"(ga) : "memory");
        }
#else
        #pragma unroll
        for (int i = 0; i < 6; ++i) {
            uint4 v = *(const uint4*)(g + i * 64);
            *(uint4*)&sbuf[sb + row * LROW + j * 8 + i * 64] = v;
        }
#endif
    };

    // ---- load & convert A fragments (16x32 bf16, 12 d-tiles), ISA layout:
    // lanes 0-15: VGPR0-3 = K 0..7, VGPR4-7 = K 16..23 (of this 32-wide tile)
    // lanes16-31: VGPR0-3 = K 8..15, VGPR4-7 = K 24..31
    v16bf a[12];
    {
        const float* xrow = x + (row0 + lr) * 384;
        #pragma unroll
        for (int dt = 0; dt < 12; ++dt) {
            const int db = dt * 32 + half * 8;
            float4 f0 = *(const float4*)(xrow + db);
            float4 f1 = *(const float4*)(xrow + db + 4);
            float4 f2 = *(const float4*)(xrow + db + 16);
            float4 f3 = *(const float4*)(xrow + db + 20);
            v8u u;
            u[0] = pk(f0.x, f0.y); u[1] = pk(f0.z, f0.w);
            u[2] = pk(f1.x, f1.y); u[3] = pk(f1.z, f1.w);
            u[4] = pk(f2.x, f2.y); u[5] = pk(f2.z, f2.w);
            u[6] = pk(f3.x, f3.y); u[7] = pk(f3.z, f3.w);
            a[dt] = __builtin_bit_cast(v16bf, u);
        }
    }

    float best[8];
    int   bidx[8];
    #pragma unroll
    for (int r = 0; r < 8; ++r) { best[r] = 3.4e38f; bidx[r] = 0; }

    const int ntiles = K >> 5;          // 32 codes per tile
    stage(0, 0);
    for (int ct = 0; ct < ntiles; ++ct) {
#if VQ_USE_ASYNC_LDS
        asm volatile("s_wait_asynccnt 0" ::: "memory");
#endif
        __syncthreads();   // tile ct visible to all; buffer (ct+1)&1 free for reuse
        if (ct + 1 < ntiles) stage((ct + 1) * 32, ((ct + 1) & 1) * SBELEM);

        const int sb = (ct & 1) * SBELEM;
        // B fragment lane layout (32x16 bf16): lane<16 holds K 0..7,16..23 of its code
        // column; lane>=16 holds K 8..15,24..31 (matches A layout).
        const int b0 = sb + lr * LROW + half * 8;         // codes cbase+0..15
        const int b1 = sb + (lr + 16) * LROW + half * 8;  // codes cbase+16..31
        v8f c0 = {}, c1 = {};
        #pragma unroll
        for (int dt = 0; dt < 12; ++dt) {
            B32 r0, r1;
            r0.lo = *(const uint4*)&sbuf[b0 + dt * 32];
            r0.hi = *(const uint4*)&sbuf[b0 + dt * 32 + 16];
            r1.lo = *(const uint4*)&sbuf[b1 + dt * 32];
            r1.hi = *(const uint4*)&sbuf[b1 + dt * 32 + 16];
            c0 = __builtin_amdgcn_wmma_f32_16x16x32_bf16(
                     false, a[dt], false, __builtin_bit_cast(v16bf, r0),
                     (short)0, c0, false, false);
            c1 = __builtin_amdgcn_wmma_f32_16x16x32_bf16(
                     false, a[dt], false, __builtin_bit_cast(v16bf, r1),
                     (short)0, c1, false, false);
        }

        const int   cbase = ct * 32;
        const float cbn0  = cbnorm[cbase + lr];
        const float cbn1  = cbnorm[cbase + 16 + lr];
        #pragma unroll
        for (int r = 0; r < 8; ++r) {
            // C layout: VGPR r, lane<16 -> M=r, N=lr ; lane>=16 -> M=8+r, N=lr
            float s0 = __builtin_fmaf(-2.0f, c0[r], cbn0);
            float s1 = __builtin_fmaf(-2.0f, c1[r], cbn1);
            if (s0 < best[r]) { best[r] = s0; bidx[r] = cbase + lr; }
            if (s1 < best[r]) { best[r] = s1; bidx[r] = cbase + 16 + lr; }
        }
    }

    // cross-lane argmin over the 16 codes per half (masks stay within halves)
    #pragma unroll
    for (int r = 0; r < 8; ++r) {
        float b_ = best[r];
        int   i_ = bidx[r];
        #pragma unroll
        for (int m = 1; m < 16; m <<= 1) {
            float ob = __shfl_xor(b_, m, 32);
            int   oi = __shfl_xor(i_, m, 32);
            if (ob < b_ || (ob == b_ && oi < i_)) { b_ = ob; i_ = oi; }
        }
        if (lr == 0)  // lanes 0 and 16 hold rows r and 8+r
            idxf_out[row0 + half * 8 + r] = (float)i_;
    }
}

// ---------- gather q = cb[idx], write q, accumulate sum((q-x)^2) ----------
__global__ __launch_bounds__(384)
void gather_loss_kernel(const float* __restrict__ x,
                        const float* __restrict__ cb,
                        const float* __restrict__ idxf,
                        float* __restrict__ q_out,
                        float* __restrict__ acc) {
    __shared__ float sw[12];
    const int row = blockIdx.x;
    const int e   = threadIdx.x;
    const int k   = (int)idxf[row];
    const float q  = cb[(size_t)k * 384 + e];
    const float xv = x[(size_t)row * 384 + e];
    const float d  = q - xv;
    q_out[(size_t)row * 384 + e] = q;

    float s = d * d;
    #pragma unroll
    for (int m = 16; m >= 1; m >>= 1) s += __shfl_xor(s, m, 32);
    if ((e & 31) == 0) sw[e >> 5] = s;
    __syncthreads();
    if (e == 0) {
        float t = 0.f;
        #pragma unroll
        for (int i = 0; i < 12; ++i) t += sw[i];
        atomicAdd(acc, t);
    }
}

// ---------- combine level sums into total vq loss ----------
__global__ void finalize_kernel(const float* __restrict__ acc,
                                float* __restrict__ out_loss) {
    if (threadIdx.x == 0) {
        out_loss[0] = 1.05f * acc[0] * (1.0f / (6272.0f   * 384.0f))
                    + 1.25f * acc[1] * (1.0f / (25088.0f  * 384.0f))
                    + 1.60f * acc[2] * (1.0f / (100352.0f * 384.0f));
    }
}

// ---------- host ----------
extern "C" void kernel_launch(void* const* d_in, const int* in_sizes, int n_in,
                              void* d_out, int out_size, void* d_ws, size_t ws_size,
                              hipStream_t stream) {
    (void)in_sizes; (void)n_in; (void)out_size; (void)ws_size;
    const float* l0  = (const float*)d_in[0];
    const float* l1  = (const float*)d_in[1];
    const float* l2  = (const float*)d_in[2];
    const float* cb0 = (const float*)d_in[3];
    const float* cb1 = (const float*)d_in[4];
    const float* cb2 = (const float*)d_in[5];
    float* out = (float*)d_out;

    const int D = 384;
    const int N0 = 32 * 196, N1 = 32 * 784, N2 = 32 * 3136;
    const int K0 = 512, K1 = 1024, K2 = 2048;

    // workspace carve-up (256B aligned)
    char* w = (char*)d_ws;
    size_t off = 0;
    auto carve = [&](size_t bytes) -> char* {
        char* p = w + off;
        off = (off + bytes + 255) & ~(size_t)255;
        return p;
    };
    unsigned short* cbf0 = (unsigned short*)carve((size_t)K0 * D * 2);
    unsigned short* cbf1 = (unsigned short*)carve((size_t)K1 * D * 2);
    unsigned short* cbf2 = (unsigned short*)carve((size_t)K2 * D * 2);
    float* nrm0 = (float*)carve((size_t)K0 * 4);
    float* nrm1 = (float*)carve((size_t)K1 * 4);
    float* nrm2 = (float*)carve((size_t)K2 * 4);
    float* acc  = (float*)carve(16);

    // d_out layout: idx0, idx1, idx2, loss, q0, q1, q2 (all as float)
    float* idxf0 = out;
    float* idxf1 = out + N0;
    float* idxf2 = out + N0 + N1;
    float* lossp = out + N0 + N1 + N2;
    float* q0    = lossp + 1;
    float* q1    = q0 + (size_t)N0 * D;
    float* q2    = q1 + (size_t)N1 * D;

    // pre-passes
    cvt_bf16_kernel<<<(K0 * D + 255) / 256, 256, 0, stream>>>(cb0, cbf0, K0 * D);
    cvt_bf16_kernel<<<(K1 * D + 255) / 256, 256, 0, stream>>>(cb1, cbf1, K1 * D);
    cvt_bf16_kernel<<<(K2 * D + 255) / 256, 256, 0, stream>>>(cb2, cbf2, K2 * D);
    norm_kernel<<<(K0 + 255) / 256, 256, 0, stream>>>(cb0, nrm0, K0);
    norm_kernel<<<(K1 + 255) / 256, 256, 0, stream>>>(cb1, nrm1, K1);
    norm_kernel<<<(K2 + 255) / 256, 256, 0, stream>>>(cb2, nrm2, K2);
    init_acc_kernel<<<1, 32, 0, stream>>>(acc);

    // WMMA argmin (N divisible by 128, K divisible by 32 for all levels)
    vq_argmin_kernel<<<N0 / 128, 256, 0, stream>>>(l0, cbf0, nrm0, idxf0, K0);
    vq_argmin_kernel<<<N1 / 128, 256, 0, stream>>>(l1, cbf1, nrm1, idxf1, K1);
    vq_argmin_kernel<<<N2 / 128, 256, 0, stream>>>(l2, cbf2, nrm2, idxf2, K2);

    // gather + loss
    gather_loss_kernel<<<N0, 384, 0, stream>>>(l0, cb0, idxf0, q0, acc + 0);
    gather_loss_kernel<<<N1, 384, 0, stream>>>(l1, cb1, idxf1, q1, acc + 1);
    gather_loss_kernel<<<N2, 384, 0, stream>>>(l2, cb2, idxf2, q2, acc + 2);

    finalize_kernel<<<1, 32, 0, stream>>>(acc, lossp);
}
